// ModifiedAttention_33165737460001
// MI455X (gfx1250) — matrix operand order repaired and checked by reading, not compiled, not run on hardware
//
#include <hip/hip_runtime.h>

// ---------------------------------------------------------------------------
// CDNA5 (gfx1250) bf16 WMMA implementation of LayerNormed-QK causal attention.
// wave32 only. Matrix math via v_wmma_f32_16x16x32_bf16; GEMM operands staged
// through LDS with (when available) the gfx1250 async-to-LDS copy engine.
// ---------------------------------------------------------------------------

typedef __bf16 bf16;
typedef __attribute__((ext_vector_type(16))) __bf16 v16bf;
typedef __attribute__((ext_vector_type(8)))  __bf16 v8bf;
typedef __attribute__((ext_vector_type(8)))  float  v8f;

union FragBF { v16bf v; v8bf h[2]; };
union AccF   { v8f   v; float f[8]; };

// ---- gfx1250 async copy to LDS (guarded; falls back to plain LDS stores) ---
#if defined(__has_builtin)
#  if __has_builtin(__builtin_amdgcn_global_load_async_to_lds_b128)
#    define HAVE_ASYNC_LDS 1
#  endif
#  if __has_builtin(__builtin_amdgcn_s_wait_asynccnt)
#    define HAVE_WAIT_ASYNC 1
#  endif
#endif

#if defined(HAVE_ASYNC_LDS)
// Builtin expects: (int __vector(4) addrspace(1)*, int __vector(4) addrspace(3)*, Imm, Imm)
typedef int v4i_gcc __attribute__((vector_size(16)));
typedef __attribute__((address_space(1))) v4i_gcc* gp16_t;
typedef __attribute__((address_space(3))) v4i_gcc* lp16_t;
static __device__ inline void cp16_g2l(const void* g, void* l) {
  __builtin_amdgcn_global_load_async_to_lds_b128((gp16_t)g, (lp16_t)l, 0, 0);
}
#  if defined(HAVE_WAIT_ASYNC)
#    define WAIT_ASYNC(n) __builtin_amdgcn_s_wait_asynccnt(n)
#  else
#    define WAIT_ASYNC(n) asm volatile("s_wait_asynccnt %0" :: "i"(n) : "memory")
#  endif
#else
static __device__ inline void cp16_g2l(const void* g, void* l) {
  *(v8bf*)l = *(const v8bf*)g;           // sync fallback; __syncthreads orders
}
#  define WAIT_ASYNC(n) ((void)0)
#endif

static __device__ inline v8f wmma_bf16(v16bf a, v16bf b, v8f c) {
  // (neg_a, A, neg_b, B, c_mod, C, reuse_a, reuse_b)
  return __builtin_amdgcn_wmma_f32_16x16x32_bf16(false, a, false, b,
                                                 (short)0, c, false, false);
}

// A fragment: 16x32 bf16, row-major source, row stride ld (elements).
// ISA layout: lane m(=l&15); lanes 0-15 hold K={0..7,16..23}, lanes 16-31 K={8..15,24..31}.
static __device__ inline v16bf load_frag_a(const bf16* base, int ld) {
  const int l  = threadIdx.x & 31;
  const int m  = l & 15;
  const int ko = (l >> 4) << 3;          // 0 or 8
  const bf16* p = base + (size_t)m * ld + ko;
  FragBF f;
  f.h[0] = *(const v8bf*)(p);            // K = ko .. ko+7
  f.h[1] = *(const v8bf*)(p + 16);       // K = 16+ko .. 16+ko+7
  return f.v;
}

// B fragment: 32x16 bf16 where the source is stored [N][K] row-major (we
// compute X @ W^T, so column n of B is row n of the source). Lane n(=l&15);
// lanes 0-15 hold K={0..7,16..23}, lanes 16-31 K={8..15,24..31}.
static __device__ inline v16bf load_frag_b(const bf16* base, int ld) {
  const int l  = threadIdx.x & 31;
  const int n  = l & 15;
  const int ko = (l >> 4) << 3;          // 0 or 8
  const bf16* p = base + (size_t)n * ld + ko;
  FragBF f;
  f.h[0] = *(const v8bf*)(p);
  f.h[1] = *(const v8bf*)(p + 16);
  return f.v;
}

// ---------------------------------------------------------------------------
// fp32 -> bf16 elementwise convert
// ---------------------------------------------------------------------------
__global__ void cvt_f32_bf16(const float* __restrict__ src,
                             bf16* __restrict__ dst, size_t n) {
  size_t i = (size_t)blockIdx.x * blockDim.x + threadIdx.x;
  if (i < n) dst[i] = (bf16)src[i];
}

// ---------------------------------------------------------------------------
// C[M,N] (f32 or bf16) = A[M,K]bf16 @ B[N,K]bf16^T (+bias for f32 mode).
// Block (256 thr = 8 waves) computes a 64x128 macro-tile. Per 32-deep K-step,
// A(64x32) and B(128x32) tiles are staged in double-buffered LDS (pitch 40 to
// avoid bank conflicts) via async-to-LDS copies, overlapped with 4 WMMAs/wave.
// Wave (wm,wn) in a 4x2 layout owns a 16x64 sub-tile.
// ---------------------------------------------------------------------------
#define APITCH 40
__global__ __launch_bounds__(256)
void gemm_bf16_nt(const bf16* __restrict__ A, const bf16* __restrict__ Bm,
                  const float* __restrict__ bias, void* __restrict__ Cout,
                  int M, int N, int K, int outF32) {
  __shared__ __align__(16) bf16 As[2][64  * APITCH];
  __shared__ __align__(16) bf16 Bs[2][128 * APITCH];

  const int tid = threadIdx.x;
  const int l   = tid & 31;
  const int wid = tid >> 5;
  const int wm  = wid & 3;               // 0..3 -> M sub-tile
  const int wn  = wid >> 2;              // 0..1 -> N sub-tile
  const int nb  = N >> 7;                // blocks along N
  const int bm  = blockIdx.x / nb;
  const int bn  = blockIdx.x % nb;

  const bf16* Ag = A  + (size_t)(bm * 64)  * K;
  const bf16* Bg = Bm + (size_t)(bn * 128) * K;

  // per-thread staging assignment: A = 1x16B chunk, B = 2x16B chunks
  const int arow = tid >> 2, aseg = tid & 3;

  AccF acc[4];
  #pragma unroll
  for (int i = 0; i < 4; ++i)
    #pragma unroll
    for (int r = 0; r < 8; ++r) acc[i].f[r] = 0.f;

  // stage tiles for k0 into LDS buffer s
  auto stage = [&](int s, int k0) {
    cp16_g2l(Ag + (size_t)arow * K + k0 + aseg * 8,
             &As[s][arow * APITCH + aseg * 8]);
    #pragma unroll
    for (int j = 0; j < 2; ++j) {
      int idx = tid + 256 * j;
      int row = idx >> 2, seg = idx & 3;
      cp16_g2l(Bg + (size_t)row * K + k0 + seg * 8,
               &Bs[s][row * APITCH + seg * 8]);
    }
  };

  stage(0, 0);
  for (int k0 = 0; k0 < K; k0 += 32) {
    const int cur = (k0 >> 5) & 1;
    if (k0 + 32 < K) {
      stage(cur ^ 1, k0 + 32);
      WAIT_ASYNC(3);                     // current buffer's 3 ops done
    } else {
      WAIT_ASYNC(0);
    }
    __syncthreads();

    v16bf af = load_frag_a(&As[cur][(wm * 16) * APITCH], APITCH);
    #pragma unroll
    for (int i = 0; i < 4; ++i) {
      v16bf bfv = load_frag_b(&Bs[cur][(wn * 64 + i * 16) * APITCH], APITCH);
      acc[i].v = wmma_bf16(af, bfv, acc[i].v);
    }
    __syncthreads();                     // done reading before next overwrite
  }

  // C/D layout: VGPR r, lane l -> row = r + 8*(l>>4), col = l&15
  const int half = l >> 4, col = l & 15;
  #pragma unroll
  for (int i = 0; i < 4; ++i)
    #pragma unroll
    for (int r = 0; r < 8; ++r) {
      int row = bm * 64  + wm * 16 + r + 8 * half;
      int cc  = bn * 128 + wn * 64 + i * 16 + col;
      float vv = acc[i].f[r];
      if (outF32) {
        if (bias) vv += bias[cc];
        ((float*)Cout)[(size_t)row * N + cc] = vv;
      } else {
        ((bf16*)Cout)[(size_t)row * N + cc] = (bf16)vv;
      }
    }
}

// ---------------------------------------------------------------------------
// In-place LayerNorm over contiguous 128-element head vectors (bf16).
// One wave per vector; 4 elems per lane; full-wave shfl_xor reductions.
// ---------------------------------------------------------------------------
__global__ __launch_bounds__(256)
void ln_head_inplace(bf16* __restrict__ q, int nvec) {
  int w = (int)((blockIdx.x * blockDim.x + threadIdx.x) >> 5);
  int l = threadIdx.x & 31;
  if (w >= nvec) return;
  bf16* p = q + (size_t)w * 128;
  float x[4];
  #pragma unroll
  for (int i = 0; i < 4; ++i) x[i] = (float)p[l * 4 + i];
  float s = x[0] + x[1] + x[2] + x[3];
  #pragma unroll
  for (int m = 1; m < 32; m <<= 1) s += __shfl_xor(s, m);
  float mu = s * (1.f / 128.f);
  float vs = 0.f;
  #pragma unroll
  for (int i = 0; i < 4; ++i) { float d = x[i] - mu; vs += d * d; }
  #pragma unroll
  for (int m = 1; m < 32; m <<= 1) vs += __shfl_xor(vs, m);
  float inv = rsqrtf(vs * (1.f / 128.f) + 1e-5f);
  #pragma unroll
  for (int i = 0; i < 4; ++i) p[l * 4 + i] = (bf16)((x[i] - mu) * inv);
}

// ---------------------------------------------------------------------------
// V [b][l][h][d] bf16 -> Vt [b][h][d][l] bf16 (makes P@V B-frags contiguous)
// ---------------------------------------------------------------------------
__global__ void transpose_v(const bf16* __restrict__ v, bf16* __restrict__ vt,
                            int B_, int L_, int H_, int hd) {
  size_t i = (size_t)blockIdx.x * blockDim.x + threadIdx.x;
  size_t n = (size_t)B_ * L_ * H_ * hd;
  if (i >= n) return;
  int d = (int)(i % hd); size_t t = i / hd;
  int h = (int)(t % H_); t /= H_;
  int lq = (int)(t % L_);
  int b  = (int)(t / L_);
  vt[(((size_t)b * H_ + h) * hd + d) * L_ + lq] = v[i];
}

// ---------------------------------------------------------------------------
// Flash-style causal attention. One wave per (b, h, 16-row q tile).
// Streams 32-key chunks: 8 WMMAs for S, fp32 online softmax (16-lane halves),
// P staged through LDS as bf16, 8 WMMAs for P@V. O accum 16x128 fp32 in VGPRs.
// ---------------------------------------------------------------------------
__global__ __launch_bounds__(32)
void attn_kernel(const bf16* __restrict__ q, const bf16* __restrict__ k,
                 const bf16* __restrict__ vt, bf16* __restrict__ out,
                 int Bsz, int L, int H) {
  __shared__ __align__(16) bf16 psh[16][40];   // 16x32 P tile, padded rows
  const int QT = L >> 4;
  const int bid = blockIdx.x;
  const int qt = bid % QT;
  const int h  = (bid / QT) % H;
  const int b  = bid / (QT * H);
  const int D  = H * 128;
  const int l = threadIdx.x & 31;
  const int half = l >> 4, col = l & 15;

  const bf16* qbase = q + (size_t)(b * L + qt * 16) * D + h * 128;
  v16bf aq[4];
  #pragma unroll
  for (int i = 0; i < 4; ++i) aq[i] = load_frag_a(qbase + 32 * i, D);

  AccF o[8];
  #pragma unroll
  for (int nn = 0; nn < 8; ++nn)
    #pragma unroll
    for (int r = 0; r < 8; ++r) o[nn].f[r] = 0.f;

  float mcur[8], lsum[8];
  #pragma unroll
  for (int r = 0; r < 8; ++r) { mcur[r] = -1e30f; lsum[r] = 0.f; }

  // scores * 1/sqrt(128), folded into base-2 exponent domain
  const float SC = 0.0883883476483184f * 1.44269504088896f;
  const int cmax = (qt * 16 + 15) >> 5;
  const bf16* vt_bh = vt + ((size_t)(b * H + h) * 128) * L;

  for (int c = 0; c <= cmax; ++c) {
    const int kb = c * 32;
    const bf16* kbase = k + (size_t)(b * L + kb) * D + h * 128;

    AccF s0, s1;
    #pragma unroll
    for (int r = 0; r < 8; ++r) { s0.f[r] = 0.f; s1.f[r] = 0.f; }
    #pragma unroll
    for (int i = 0; i < 4; ++i) {
      v16bf bk0 = load_frag_b(kbase + 32 * i, D);
      v16bf bk1 = load_frag_b(kbase + (size_t)16 * D + 32 * i, D);
      s0.v = wmma_bf16(aq[i], bk0, s0.v);
      s1.v = wmma_bf16(aq[i], bk1, s1.v);
    }

    float e0[8], e1[8], al[8];
    #pragma unroll
    for (int r = 0; r < 8; ++r) {
      int row = qt * 16 + r + 8 * half;
      float v0 = (kb + col      <= row) ? s0.f[r] * SC : -1e30f;
      float v1 = (kb + 16 + col <= row) ? s1.f[r] * SC : -1e30f;
      float mx = fmaxf(v0, v1);
      mx = fmaxf(mx, __shfl_xor(mx, 1));
      mx = fmaxf(mx, __shfl_xor(mx, 2));
      mx = fmaxf(mx, __shfl_xor(mx, 4));
      mx = fmaxf(mx, __shfl_xor(mx, 8));
      float mnew = fmaxf(mcur[r], mx);
      float a  = exp2f(mcur[r] - mnew);
      float x0 = exp2f(v0 - mnew);
      float x1 = exp2f(v1 - mnew);
      float rs = x0 + x1;
      rs += __shfl_xor(rs, 1);
      rs += __shfl_xor(rs, 2);
      rs += __shfl_xor(rs, 4);
      rs += __shfl_xor(rs, 8);
      lsum[r] = lsum[r] * a + rs;
      mcur[r] = mnew;
      e0[r] = x0; e1[r] = x1; al[r] = a;
    }
    #pragma unroll
    for (int nn = 0; nn < 8; ++nn)
      #pragma unroll
      for (int r = 0; r < 8; ++r) o[nn].f[r] *= al[r];

    __syncthreads();                     // WAR on LDS tile (single wave)
    #pragma unroll
    for (int r = 0; r < 8; ++r) {
      psh[r + 8 * half][col]      = (bf16)e0[r];
      psh[r + 8 * half][16 + col] = (bf16)e1[r];
    }
    __syncthreads();

    v16bf ap = load_frag_a(&psh[0][0], 40);
    #pragma unroll
    for (int nn = 0; nn < 8; ++nn) {
      v16bf bv = load_frag_b(vt_bh + (size_t)(nn * 16) * L + kb, L);
      o[nn].v = wmma_bf16(ap, bv, o[nn].v);
    }
  }

  bf16* obase = out + (size_t)(b * L) * D + h * 128;
  #pragma unroll
  for (int nn = 0; nn < 8; ++nn)
    #pragma unroll
    for (int r = 0; r < 8; ++r) {
      int row = qt * 16 + r + 8 * half;
      obase[(size_t)row * D + nn * 16 + col] = (bf16)(o[nn].f[r] / lsum[r]);
    }
}

// ---------------------------------------------------------------------------
extern "C" void kernel_launch(void* const* d_in, const int* in_sizes, int n_in,
                              void* d_out, int out_size, void* d_ws, size_t ws_size,
                              hipStream_t stream) {
  (void)in_sizes; (void)n_in; (void)out_size; (void)ws_size;
  const float* x  = (const float*)d_in[0];
  const float* Wq = (const float*)d_in[1];
  const float* Wk = (const float*)d_in[2];
  const float* Wv = (const float*)d_in[3];
  const float* Wo = (const float*)d_in[4];
  const float* bo = (const float*)d_in[5];
  float* out = (float*)d_out;

  const int B = 2, L = 2048, D = 2048, H = 16;
  const size_t MD = (size_t)B * L * D;   // 8,388,608 activation elems
  const size_t WD = (size_t)D * D;       // 4,194,304 weight elems

  // Workspace layout (bf16), total = 128 MiB
  char* p = (char*)d_ws;
  bf16* xb  = (bf16*)p; p += MD * 2;
  bf16* wqb = (bf16*)p; p += WD * 2;
  bf16* wkb = (bf16*)p; p += WD * 2;
  bf16* wvb = (bf16*)p; p += WD * 2;
  bf16* wob = (bf16*)p; p += WD * 2;
  bf16* qb  = (bf16*)p; p += MD * 2;
  bf16* kb  = (bf16*)p; p += MD * 2;
  bf16* vb  = (bf16*)p; p += MD * 2;
  bf16* vtb = (bf16*)p; p += MD * 2;
  bf16* at  = (bf16*)p; p += MD * 2;

  // 1) fp32 -> bf16 converts
  cvt_f32_bf16<<<(int)((MD + 255) / 256), 256, 0, stream>>>(x,  xb,  MD);
  cvt_f32_bf16<<<(int)((WD + 255) / 256), 256, 0, stream>>>(Wq, wqb, WD);
  cvt_f32_bf16<<<(int)((WD + 255) / 256), 256, 0, stream>>>(Wk, wkb, WD);
  cvt_f32_bf16<<<(int)((WD + 255) / 256), 256, 0, stream>>>(Wv, wvb, WD);
  cvt_f32_bf16<<<(int)((WD + 255) / 256), 256, 0, stream>>>(Wo, wob, WD);

  // 2) QKV projections: [4096,2048] = xb @ W^T  (bf16 out)
  const int M = B * L, N = D, K = D;
  const int gemmBlocks = (M / 64) * (N / 128);            // 64x128 macro-tiles
  gemm_bf16_nt<<<gemmBlocks, 256, 0, stream>>>(xb, wqb, nullptr, qb, M, N, K, 0);
  gemm_bf16_nt<<<gemmBlocks, 256, 0, stream>>>(xb, wkb, nullptr, kb, M, N, K, 0);
  gemm_bf16_nt<<<gemmBlocks, 256, 0, stream>>>(xb, wvb, nullptr, vb, M, N, K, 0);

  // 3) per-head LayerNorm on q, k (in place)
  const int nvec = B * L * H;                              // 65536 vectors
  ln_head_inplace<<<nvec / 8, 256, 0, stream>>>(qb, nvec);
  ln_head_inplace<<<nvec / 8, 256, 0, stream>>>(kb, nvec);

  // 4) V -> Vt [b][h][d][l]
  transpose_v<<<(int)((MD + 255) / 256), 256, 0, stream>>>(vb, vtb, B, L, H, 128);

  // 5) causal flash attention, one wave per (b,h,qtile)
  attn_kernel<<<B * H * (L / 16), 32, 0, stream>>>(qb, kb, vtb, at, B, L, H);

  // 6) output projection with bias (fp32 out)
  gemm_bf16_nt<<<gemmBlocks, 256, 0, stream>>>(at, wob, bo, out, M, N, K, 1);
}
